// TripletLossHead_68075231642277
// MI455X (gfx1250) — matrix operand order repaired and checked by reading, not compiled, not run on hardware
//
#include <hip/hip_runtime.h>
#include <math.h>

typedef __attribute__((ext_vector_type(16))) __bf16 v16bf;
typedef __attribute__((ext_vector_type(8)))  float  v8f;

#define N_ROWS 8192
#define DIM    256
#define HID    64
#define CHUNK  512          // columns per LDS chunk in the fused dist/topk kernel
#define NTILE  (CHUNK/16)   // 32 column tiles per chunk
#define MARGIN 100.0f
#define GAMMA  2.0f

// ---------- helpers ----------

__device__ __forceinline__ unsigned short f32_to_bf16_rne(float f) {
    unsigned int u = __float_as_uint(f);
    unsigned int r = u + 0x7FFFu + ((u >> 16) & 1u);
    return (unsigned short)(r >> 16);
}

__device__ __forceinline__ float wave_reduce_sum(float v) {
    #pragma unroll
    for (int o = 16; o > 0; o >>= 1) v += __shfl_xor(v, o, 32);
    return v;
}

union LoadU { uint4 q[2]; v16bf v; };

// ---------- kernel 1: zero the loss accumulator ----------

__global__ void k_zero(float* accum) { accum[0] = 0.0f; }

// ---------- kernel 2: L2-normalize rows of a [N,256] f32 matrix -> bf16 ----------

__global__ void k_normalize_rows(const float* __restrict__ src,
                                 unsigned short* __restrict__ dst) {
    __shared__ float red[8];
    __shared__ float s_inv;
    const int row = blockIdx.x;
    const int t   = threadIdx.x;          // 256 threads, one element each
    const float v = src[row * DIM + t];

    float ss = wave_reduce_sum(v * v);
    if ((t & 31) == 0) red[t >> 5] = ss;
    __syncthreads();
    if (t == 0) {
        float s = 0.0f;
        #pragma unroll
        for (int i = 0; i < 8; ++i) s += red[i];
        s_inv = 1.0f / fmaxf(sqrtf(s), 1e-12f);
    }
    __syncthreads();
    dst[row * DIM + t] = f32_to_bf16_rne(v * s_inv);
}

// ---------- kernel 3: predictor MLP (Linear->ReLU->Linear) + L2 normalize -> bf16 ----------

__global__ void k_predictor(const float* __restrict__ input,
                            const float* __restrict__ W1,
                            const float* __restrict__ b1,
                            const float* __restrict__ W2,
                            const float* __restrict__ b2,
                            unsigned short* __restrict__ predn) {
    __shared__ float inrow[DIM];
    __shared__ float hpart[4][HID];
    __shared__ float hid[HID];
    __shared__ float red[8];
    __shared__ float s_inv;

    const int row = blockIdx.x;
    const int t   = threadIdx.x;          // 256 threads

    inrow[t] = input[row * DIM + t];
    __syncthreads();

    // hidden[j] = relu(b1[j] + sum_k inrow[k] * W1[k][j]) ; W1 is [256][64]
    const int j    = t & 63;
    const int part = t >> 6;              // 4 partial slices of K
    float acc = 0.0f;
    const int k0 = part * 64;
    #pragma unroll 8
    for (int k = 0; k < 64; ++k)
        acc = fmaf(inrow[k0 + k], W1[(k0 + k) * HID + j], acc);
    hpart[part][j] = acc;
    __syncthreads();
    if (t < HID) {
        float h = b1[t] + hpart[0][t] + hpart[1][t] + hpart[2][t] + hpart[3][t];
        hid[t] = fmaxf(h, 0.0f);
    }
    __syncthreads();

    // pred[t] = b2[t] + sum_j hid[j] * W2[j][t] ; W2 is [64][256]
    float p = b2[t];
    #pragma unroll 8
    for (int jj = 0; jj < HID; ++jj)
        p = fmaf(hid[jj], W2[jj * DIM + t], p);

    float ss = wave_reduce_sum(p * p);
    if ((t & 31) == 0) red[t >> 5] = ss;
    __syncthreads();
    if (t == 0) {
        float s = 0.0f;
        #pragma unroll
        for (int i = 0; i < 8; ++i) s += red[i];
        s_inv = 1.0f / fmaxf(sqrtf(s), 1e-12f);
    }
    __syncthreads();
    predn[row * DIM + t] = f32_to_bf16_rne(p * s_inv);
}

// ---------- kernel 4: fused dist GEMM (WMMA bf16) + streaming per-row top-10 ----------
// Block = 256 threads = 8 waves. Each block owns 16 pred rows, sweeps all 8192 columns
// in chunks of CHUNK, keeping dist tiles in LDS only.

__global__ void k_dist_topk(const unsigned short* __restrict__ predn,
                            const unsigned short* __restrict__ targn,
                            float* __restrict__ accum) {
    extern __shared__ float smem[];
    float* sdist = smem;                    // [16][CHUNK]
    float* sdiag = smem + 16 * CHUNK;       // [16]
    float* scand = sdiag + 16;              // [16][16*10]

    const int t    = threadIdx.x;
    const int wave = t >> 5;
    const int lane = t & 31;
    const bool hi  = lane >= 16;
    const int  ln  = lane & 15;
    const int row0 = blockIdx.x * 16;

    // --- load A fragments once: 16 rows x 256 K of pred_norm (bf16) ---
    // CDNA5 16-bit A layout: lanes 0-15 -> K {base+0..7, base+16..23},
    //                        lanes 16-31 -> K {base+8..15, base+24..31}
    v16bf afrag[8];
    const unsigned short* aptr = predn + (row0 + ln) * DIM;
    #pragma unroll
    for (int ks = 0; ks < 8; ++ks) {
        LoadU u;
        u.q[0] = *(const uint4*)(aptr + ks * 32 + (hi ? 8 : 0));
        u.q[1] = *(const uint4*)(aptr + ks * 32 + (hi ? 24 : 16));
        afrag[ks] = u.v;
    }

    // --- per-thread sorted top-10 (ascending), fully register resident ---
    float tk[10];
    #pragma unroll
    for (int i = 0; i < 10; ++i) tk[i] = INFINITY;

    const int srow = t >> 4;        // row this thread scans (0..15)
    const int scol = t & 15;        // column residue class mod 16
    const int g    = row0 + srow;   // global row -> diagonal column index

    for (int cb = 0; cb < N_ROWS; cb += CHUNK) {
        // ---- compute phase: each wave does NTILE/8 column tiles ----
        for (int tt = wave; tt < NTILE; tt += 8) {
            const int j0 = cb + tt * 16;
            // B layout (32x16, 16-bit): lanes 0-15 -> K {base+0..7, base+16..23}, lanes 16-31 -> K {base+8..15, base+24..31},
            // N = lane&15. B[k][n] = target_norm[j0+n][k] (same K order as A).
            const unsigned short* bptr = targn + (j0 + ln) * DIM;
            v8f c;
            #pragma unroll
            for (int r = 0; r < 8; ++r) c[r] = 0.0f;
            #pragma unroll
            for (int ks = 0; ks < 8; ++ks) {
                LoadU u;
                u.q[0] = *(const uint4*)(bptr + ks * 32 + (hi ? 8 : 0));
                u.q[1] = *(const uint4*)(bptr + ks * 32 + (hi ? 24 : 16));
                c = __builtin_amdgcn_wmma_f32_16x16x32_bf16(
                        false, afrag[ks], false, u.v, (short)0, c, false, false);
            }
            // D layout: VGPR r -> M = r + 8*hi ; N = lane&15
            const int colbase = tt * 16 + ln;
            #pragma unroll
            for (int r = 0; r < 8; ++r) {
                const int m = r + (hi ? 8 : 0);
                sdist[m * CHUNK + colbase] = -2.0f * c[r];
            }
        }
        __syncthreads();

        // ---- scan phase: maintain per-(row, residue) top-10, capture diagonal ----
        for (int c = scol; c < CHUNK; c += 16) {
            const int jg = cb + c;
            const float v = sdist[srow * CHUNK + c];
            if (jg == g) {
                sdiag[srow] = v;            // diagonal: exclude from top-k
            } else if (v < tk[9]) {
                tk[9] = v;
                #pragma unroll
                for (int i = 9; i > 0; --i) {
                    if (tk[i] < tk[i - 1]) {
                        float tmp = tk[i - 1]; tk[i - 1] = tk[i]; tk[i] = tmp;
                    }
                }
            }
        }
        __syncthreads();
    }

    // ---- publish candidates: 16 residues x 10 per row ----
    #pragma unroll
    for (int i = 0; i < 10; ++i)
        scand[srow * 160 + scol * 10 + i] = tk[i];
    __syncthreads();

    // ---- per-row final selection of 10 smallest of 160; mean of ranks 1..9 ----
    if (t < 16) {
        float* cand = scand + t * 160;
        float sum9 = 0.0f;
        for (int p = 0; p < 10; ++p) {
            float mn = INFINITY; int mi = 0;
            for (int q = 0; q < 160; ++q) {
                float v = cand[q];
                if (v < mn) { mn = v; mi = q; }
            }
            cand[mi] = INFINITY;
            if (p > 0) sum9 += mn;          // drop the smallest, keep ranks 1..9
        }
        const float dist_an = sum9 * (1.0f / 9.0f);
        const float dist_ap = sdiag[t];
        const float li = fmaxf(0.0f, GAMMA * dist_ap - dist_an + MARGIN);
        atomicAdd(accum, li);
    }
}

// ---------- kernel 5: finalize mean ----------

__global__ void k_finalize(const float* accum, float* out) {
    out[0] = accum[0] * (1.0f / (float)N_ROWS);
}

// ---------- launch ----------

extern "C" void kernel_launch(void* const* d_in, const int* in_sizes, int n_in,
                              void* d_out, int out_size, void* d_ws, size_t ws_size,
                              hipStream_t stream) {
    const float* input  = (const float*)d_in[0];
    const float* target = (const float*)d_in[1];
    const float* W1     = (const float*)d_in[2];
    const float* b1     = (const float*)d_in[3];
    const float* W2     = (const float*)d_in[4];
    const float* b2     = (const float*)d_in[5];
    float* out = (float*)d_out;

    // workspace layout
    float* accum = (float*)d_ws;
    unsigned short* predn = (unsigned short*)((char*)d_ws + 256);
    unsigned short* targn = predn + (size_t)N_ROWS * DIM;

    k_zero<<<1, 1, 0, stream>>>(accum);
    k_normalize_rows<<<N_ROWS, 256, 0, stream>>>(target, targn);
    k_predictor<<<N_ROWS, 256, 0, stream>>>(input, W1, b1, W2, b2, predn);

    const size_t smem_bytes = (size_t)(16 * CHUNK + 16 + 16 * 160) * sizeof(float);
    k_dist_topk<<<N_ROWS / 16, 256, smem_bytes, stream>>>(predn, targn, accum);

    k_finalize<<<1, 1, 0, stream>>>(accum, out);
}
